// DecoderLayer_56891136803395
// MI455X (gfx1250) — matrix operand order repaired and checked by reading, not compiled, not run on hardware
//
#include <hip/hip_runtime.h>

// ---------------------------------------------------------------------------
// Decoder layer for MI455X (gfx1250): bf16 WMMA GEMMs + flash attention.
// B=2, S=2048, D=1024, H=16, DK=64, DFF=4096.
// Data path: weights pre-transposed to [N][K] so every GEMM tile is a
// contiguous 2D tile -> loaded by the Tensor Data Mover (tensor_load_to_lds);
// V projection written transposed so attention's V^T tile is also a
// contiguous copy.  All matrix math via v_wmma_f32_16x16x32_bf16.
// ---------------------------------------------------------------------------

#define B_   2
#define S_   2048
#define D_   1024
#define H_   16
#define DK_  64
#define DFF_ 4096

typedef __bf16 bf16;
typedef __bf16 v16bf __attribute__((ext_vector_type(16)));
typedef float  v8f   __attribute__((ext_vector_type(8)));
typedef unsigned int u32x4 __attribute__((ext_vector_type(4)));
typedef int  i32x4 __attribute__((ext_vector_type(4)));
typedef int  i32x8 __attribute__((ext_vector_type(8)));

#if __has_builtin(__builtin_amdgcn_tensor_load_to_lds) && \
    __has_builtin(__builtin_amdgcn_s_wait_tensorcnt)
#define USE_TDM 1
#else
#define USE_TDM 0
#endif

union FragB {
  v16bf v;
  uint4 q[2];
  unsigned short us[16];
};

__device__ __forceinline__ bf16 f2bf(float f) {
  unsigned u = __builtin_bit_cast(unsigned, f);
  u += 0x7FFFu + ((u >> 16) & 1u);  // round-to-nearest-even
  unsigned short s = (unsigned short)(u >> 16);
  return __builtin_bit_cast(bf16, s);
}

#if USE_TDM
// Issue one TDM 2D tile load: tile [tileD1 rows][tileD0 elems] of bf16 from a
// row-major tensor with row stride strideElems, into LDS at ldsOff (linear).
// D# bit layout per CDNA5 ISA ch.8 (group0: count/lds/global/type,
// group1: data_size, dims, tile dims, dim0 stride).  6-arg builtin form:
// (g0 u32x4, g1 i32x8, g2 i32x4, g3 i32x4, i32x8, cpol).
__device__ __forceinline__ void tdm_load_2d(unsigned ldsOff, const void* gptr,
                                            unsigned tileD0, unsigned tileD1,
                                            unsigned tenD0, unsigned tenD1,
                                            unsigned strideElems) {
  unsigned long long ga = (unsigned long long)gptr;
  u32x4 g0;
  g0[0] = 1u;                                    // count=1 (user descriptor)
  g0[1] = ldsOff;                                // lds_addr
  g0[2] = (unsigned)(ga & 0xFFFFFFFFu);          // global_addr[31:0]
  g0[3] = (unsigned)((ga >> 32) & 0x1FFFFFFu)    // global_addr[56:32]
          | (2u << 30);                          // type=2 ("image")
  i32x8 g1;
  g1[0] = (int)(1u << 16);                       // data_size=1 -> 2 bytes
  g1[1] = (int)((tenD0 & 0xFFFFu) << 16);        // tensor_dim0[15:0]
  g1[2] = (int)((tenD0 >> 16) | ((tenD1 & 0xFFFFu) << 16));  // d0 hi | d1 lo
  g1[3] = (int)((tenD1 >> 16) | (tileD0 << 16)); // d1 hi | tile_dim0
  g1[4] = (int)(tileD1 & 0xFFFFu);               // tile_dim1 (tile_dim2=0)
  g1[5] = (int)strideElems;                      // tensor_dim0_stride[31:0]
  g1[6] = 0;                                     // stride hi (0), dim1_stride
  g1[7] = 0;
  i32x4 z4 = {0, 0, 0, 0};                       // groups 2/3: unused (2D)
  i32x8 z8 = {0, 0, 0, 0, 0, 0, 0, 0};
  __builtin_amdgcn_tensor_load_to_lds(g0, g1, z4, z4, z8, 0);
}
#endif

// ---------------------------------------------------------------------------
// fp32 -> bf16 elementwise conversion (grid-stride)
// ---------------------------------------------------------------------------
__global__ void cvt_f32_bf16(const float* __restrict__ in,
                             bf16* __restrict__ out, int n) {
  int i = blockIdx.x * blockDim.x + threadIdx.x;
  int stride = gridDim.x * blockDim.x;
  for (; i < n; i += stride) out[i] = f2bf(in[i]);
}

// ---------------------------------------------------------------------------
// fp32 [R][C] -> bf16 transposed [C][R], 32x32 tiles via LDS.
// ---------------------------------------------------------------------------
__global__ __launch_bounds__(256) void cvt_transpose_bf16(
    const float* __restrict__ in, bf16* __restrict__ out, int R, int C) {
  __shared__ float t[32][33];
  const int c0 = blockIdx.x * 32;
  const int r0 = blockIdx.y * 32;
  const int tx = threadIdx.x & 31;
  const int ty = threadIdx.x >> 5;  // 0..7
#pragma unroll
  for (int j = 0; j < 4; j++)
    t[ty + j * 8][tx] = in[(size_t)(r0 + ty + j * 8) * C + c0 + tx];
  __syncthreads();
#pragma unroll
  for (int j = 0; j < 4; j++)
    out[(size_t)(c0 + ty + j * 8) * R + r0 + tx] = f2bf(t[tx][ty + j * 8]);
}

// ---------------------------------------------------------------------------
// GEMM: C[M,N] = act(A[M,K] @ W[K,N] + bias[N]) with W given TRANSPOSED as
// Wt[N][K].  256 threads = 8 waves; block tile 128x128; K-step 32.
// Wave (wm 0..3, wn 0..1) owns 32x64 = 2x4 WMMA accumulators.
// Both LDS tiles are [row][k] pitch-32, so A and B fragments are each two
// contiguous b128 LDS loads matching the 16-bit WMMA A/B VGPR layouts.
// Tiles are fetched by the Tensor Data Mover (wave 0 issues 2 descriptors per
// K-step, s_wait_tensorcnt, barrier) when available.
// Output modes (templates, no runtime branches):
//   WF: fp32 [M][N];  WB: bf16 [M][N];  VT: bf16 transposed [B][H][DK][S].
// ---------------------------------------------------------------------------
template <int RELU, int WF, int WB, int VT>
__global__ __launch_bounds__(256) void gemm_bf16_wmma(
    const bf16* __restrict__ A, const bf16* __restrict__ Wt,
    const float* __restrict__ bias, float* __restrict__ outF,
    bf16* __restrict__ outB, int M, int N, int K) {
  __shared__ bf16 As[128 * 32];
  __shared__ bf16 Bs[128 * 32];

  const int tid  = threadIdx.x;
  const int lane = tid & 31;
  const int wid  = tid >> 5;
  const int half = lane >> 4;
  const int l16  = lane & 15;
  const int m0   = blockIdx.y * 128;
  const int n0   = blockIdx.x * 128;
  const int wm   = (wid & 3) * 32;
  const int wn   = (wid >> 2) * 64;

  v8f acc[2][4];
#pragma unroll
  for (int mi = 0; mi < 2; mi++)
#pragma unroll
    for (int ni = 0; ni < 4; ni++)
#pragma unroll
      for (int e = 0; e < 8; e++) acc[mi][ni][e] = 0.0f;

#if USE_TDM
  const unsigned asOff = (unsigned)(size_t)(void*)As;
  const unsigned bsOff = (unsigned)(size_t)(void*)Bs;
#endif

  for (int k0 = 0; k0 < K; k0 += 32) {
#if USE_TDM
    if (wid == 0) {
      tdm_load_2d(asOff, &A[(size_t)m0 * K + k0], 32u, 128u, (unsigned)K,
                  (unsigned)M, (unsigned)K);
      tdm_load_2d(bsOff, &Wt[(size_t)n0 * K + k0], 32u, 128u, (unsigned)K,
                  (unsigned)N, (unsigned)K);
      __builtin_amdgcn_s_wait_tensorcnt(0);
    }
    __syncthreads();
#else
    {  // cooperative b128 copies: 16 bf16 per thread per tile
      int row = tid >> 1;
      int cb  = (tid & 1) * 16;
      const bf16* ga = &A[(size_t)(m0 + row) * K + k0 + cb];
      uint4 a0 = *(const uint4*)(ga);
      uint4 a1 = *(const uint4*)(ga + 8);
      *(uint4*)&As[row * 32 + cb]     = a0;
      *(uint4*)&As[row * 32 + cb + 8] = a1;
      const bf16* gb = &Wt[(size_t)(n0 + row) * K + k0 + cb];
      uint4 b0 = *(const uint4*)(gb);
      uint4 b1 = *(const uint4*)(gb + 8);
      *(uint4*)&Bs[row * 32 + cb]     = b0;
      *(uint4*)&Bs[row * 32 + cb + 8] = b1;
      if (k0 + 32 < K) {
        __builtin_prefetch((const void*)(ga + 32), 0, 1);
        __builtin_prefetch((const void*)(gb + 32), 0, 1);
      }
    }
    __syncthreads();
#endif

    FragB a[2], b[4];
#pragma unroll
    for (int mi = 0; mi < 2; mi++) {
      // A frag: lane(m=l16): e0..7 -> k=half*8+0..7 ; e8..15 -> 16+half*8+0..7
      const bf16* p = &As[(wm + mi * 16 + l16) * 32 + half * 8];
      a[mi].q[0] = *(const uint4*)(p);
      a[mi].q[1] = *(const uint4*)(p + 16);
    }
#pragma unroll
    for (int ni = 0; ni < 4; ni++) {
      // B frag: lane(n=l16): e0..7 -> k=half*8+0..7 ; e8..15 -> 16+half*8+0..7
      const bf16* p = &Bs[(wn + ni * 16 + l16) * 32 + half * 8];
      b[ni].q[0] = *(const uint4*)(p);
      b[ni].q[1] = *(const uint4*)(p + 16);
    }
#pragma unroll
    for (int mi = 0; mi < 2; mi++)
#pragma unroll
      for (int ni = 0; ni < 4; ni++)
        acc[mi][ni] = __builtin_amdgcn_wmma_f32_16x16x32_bf16(
            false, a[mi].v, false, b[ni].v, (short)0, acc[mi][ni], false,
            false);
    __syncthreads();
  }

  // Epilogue: C layout -> row = base + e + half*8, col = base + l16
#pragma unroll
  for (int mi = 0; mi < 2; mi++)
#pragma unroll
    for (int ni = 0; ni < 4; ni++) {
      int col  = n0 + wn + ni * 16 + l16;
      float bi = bias[col];
#pragma unroll
      for (int e = 0; e < 8; e++) {
        int row = m0 + wm + mi * 16 + e + half * 8;
        float v = acc[mi][ni][e] + bi;
        if (RELU) v = v > 0.0f ? v : 0.0f;
        if (WF) outF[(size_t)row * N + col] = v;
        if (WB) outB[(size_t)row * N + col] = f2bf(v);
        if (VT) {  // bf16, [B][H][DK][S] layout (M must equal B_*S_)
          int rb = row >> 11, rs = row & (S_ - 1);
          int hh = col >> 6,  dk = col & (DK_ - 1);
          outB[(((size_t)rb * H_ + hh) * DK_ + dk) * S_ + rs] = f2bf(v);
        }
      }
    }
}

// ---------------------------------------------------------------------------
// Flash attention: one block = (64 q rows, one head, one batch); 4 waves,
// each wave owns 16 q rows; KV blocks of 32 with online softmax.
// Q/K in GEMM layout [b*S+s, h*DK+dk]; V pre-transposed [B][H][DK][S] so the
// V^T tile is a contiguous copy.  Causal mask computed analytically.
// ---------------------------------------------------------------------------
template <int CAUSAL>
__global__ __launch_bounds__(128) void attn_wmma(
    const bf16* __restrict__ Qb, const bf16* __restrict__ Kb,
    const bf16* __restrict__ Vtg, bf16* __restrict__ Ob, int Sq, int Sk) {
  __shared__ bf16 Ks[32 * 64];  // [kv][dk]
  __shared__ bf16 Vt[64 * 32];  // [dk][kv]
  __shared__ bf16 Pst[4][16 * 32];

  const int tid  = threadIdx.x;
  const int lane = tid & 31;
  const int wid  = tid >> 5;
  const int half = lane >> 4;
  const int l16  = lane & 15;
  const int h    = blockIdx.y;
  const int bb   = blockIdx.z;
  const int q0   = blockIdx.x * 64 + wid * 16;
  const float scale = 0.125f;  // 1/sqrt(64)

  // Q fragments (A layout) straight from global: lane reads row q0+l16
  FragB qa[2];
  {
    const bf16* qp = &Qb[((size_t)bb * Sq + (q0 + l16)) * D_ + h * DK_];
#pragma unroll
    for (int kc = 0; kc < 2; kc++) {
      qa[kc].q[0] = *(const uint4*)(qp + kc * 32 + half * 8);
      qa[kc].q[1] = *(const uint4*)(qp + kc * 32 + 16 + half * 8);
    }
  }

  v8f o[4];
  float mrow[8], lrow[8];
#pragma unroll
  for (int ni = 0; ni < 4; ni++)
#pragma unroll
    for (int e = 0; e < 8; e++) o[ni][e] = 0.0f;
#pragma unroll
  for (int e = 0; e < 8; e++) { mrow[e] = -1e30f; lrow[e] = 0.0f; }

  int kvend = Sk;
  if (CAUSAL) {
    int lim = blockIdx.x * 64 + 64;
    kvend   = lim < Sk ? lim : Sk;
  }

  for (int kv0 = 0; kv0 < kvend; kv0 += 32) {
    {  // K tile [32 kv][64 dk]: contiguous b128 copies
      int row = tid >> 2;
      int cb  = (tid & 3) * 16;
      const bf16* kg = &Kb[((size_t)bb * Sk + kv0 + row) * D_ + h * DK_ + cb];
      uint4 k0q = *(const uint4*)(kg);
      uint4 k1q = *(const uint4*)(kg + 8);
      *(uint4*)&Ks[row * 64 + cb]     = k0q;
      *(uint4*)&Ks[row * 64 + cb + 8] = k1q;
      if (kv0 + 32 < kvend)
        __builtin_prefetch((const void*)(kg + (size_t)32 * D_), 0, 1);
    }
    {  // V^T tile [64 dk][32 kv]: contiguous b128 copies from [B][H][DK][S]
      int row = tid >> 1;        // dk 0..63
      int cb  = (tid & 1) * 16;  // kv 0/16
      const bf16* vg =
          &Vtg[(((size_t)bb * H_ + h) * DK_ + row) * Sk + kv0 + cb];
      uint4 v0q = *(const uint4*)(vg);
      uint4 v1q = *(const uint4*)(vg + 8);
      *(uint4*)&Vt[row * 32 + cb]     = v0q;
      *(uint4*)&Vt[row * 32 + cb + 8] = v1q;
      if (kv0 + 32 < kvend) __builtin_prefetch((const void*)(vg + 32), 0, 1);
    }
    __syncthreads();

    // scores S[16q, 32kv] as two 16x16 C tiles; dk contracted in 2 WMMAs
    v8f s[2];
#pragma unroll
    for (int n = 0; n < 2; n++) {
#pragma unroll
      for (int e = 0; e < 8; e++) s[n][e] = 0.0f;
#pragma unroll
      for (int kc = 0; kc < 2; kc++) {
        FragB bfr;  // B frag: lane = key column; elements = dk in table order
        const bf16* p = &Ks[(n * 16 + l16) * 64 + kc * 32 + half * 8];
        bfr.q[0] = *(const uint4*)(p);
        bfr.q[1] = *(const uint4*)(p + 16);
        s[n] = __builtin_amdgcn_wmma_f32_16x16x32_bf16(
            false, qa[kc].v, false, bfr.v, (short)0, s[n], false, false);
      }
    }

    // scale + causal mask (row = e + half*8, col = n*16 + l16)
#pragma unroll
    for (int n = 0; n < 2; n++)
#pragma unroll
      for (int e = 0; e < 8; e++) {
        float v = s[n][e] * scale;
        if (CAUSAL) {
          int qg = q0 + e + half * 8;
          int kg = kv0 + n * 16 + l16;
          if (kg > qg) v = -1e30f;
        }
        s[n][e] = v;
      }

    // online softmax: row reductions across the 16-lane group holding the row
    float p0[8], p1[8], alpha[8];
#pragma unroll
    for (int e = 0; e < 8; e++) {
      float mx = fmaxf(s[0][e], s[1][e]);
#pragma unroll
      for (int off = 1; off < 16; off <<= 1)
        mx = fmaxf(mx, __shfl_xor(mx, off, 32));
      float mnew = fmaxf(mrow[e], mx);
      float a    = __expf(mrow[e] - mnew);
      float e0   = __expf(s[0][e] - mnew);
      float e1   = __expf(s[1][e] - mnew);
      float rs   = e0 + e1;
#pragma unroll
      for (int off = 1; off < 16; off <<= 1) rs += __shfl_xor(rs, off, 32);
      lrow[e]  = lrow[e] * a + rs;
      mrow[e]  = mnew;
      alpha[e] = a;
      p0[e] = e0;
      p1[e] = e1;
    }
#pragma unroll
    for (int ni = 0; ni < 4; ni++)
#pragma unroll
      for (int e = 0; e < 8; e++) o[ni][e] *= alpha[e];

    // restage P (C layout) into LDS, read back as A fragment
#pragma unroll
    for (int e = 0; e < 8; e++) {
      int r = e + half * 8;
      Pst[wid][r * 32 + l16]      = f2bf(p0[e]);
      Pst[wid][r * 32 + 16 + l16] = f2bf(p1[e]);
    }
    __syncthreads();
    FragB pf;
    {
      const bf16* p = &Pst[wid][l16 * 32 + half * 8];
      pf.q[0] = *(const uint4*)(p);
      pf.q[1] = *(const uint4*)(p + 16);
    }
    // O[16q,64dk] += P @ V  (4 WMMAs over the dk column tiles)
#pragma unroll
    for (int ni = 0; ni < 4; ni++) {
      FragB vfr;
      const bf16* p = &Vt[(ni * 16 + l16) * 32 + half * 8];
      vfr.q[0] = *(const uint4*)(p);
      vfr.q[1] = *(const uint4*)(p + 16);
      o[ni] = __builtin_amdgcn_wmma_f32_16x16x32_bf16(
          false, pf.v, false, vfr.v, (short)0, o[ni], false, false);
    }
    __syncthreads();
  }

  // normalize + store bf16 (feeds output-projection GEMM)
#pragma unroll
  for (int e = 0; e < 8; e++) {
    float inv = 1.0f / lrow[e];
    int qg    = q0 + e + half * 8;
    bf16* op  = &Ob[((size_t)bb * Sq + qg) * D_ + h * DK_];
#pragma unroll
    for (int ni = 0; ni < 4; ni++) op[ni * 16 + l16] = f2bf(o[ni][e] * inv);
  }
}

// ---------------------------------------------------------------------------
// Fused residual add + LayerNorm over D=1024. One block per row.
// ---------------------------------------------------------------------------
__global__ __launch_bounds__(256) void add_ln(
    const float* __restrict__ x, const float* __restrict__ r,
    const float* __restrict__ g, const float* __restrict__ bta,
    float* __restrict__ outF, bf16* __restrict__ outB) {
  __shared__ float red[8];
  const int row = blockIdx.x;
  const int tid = threadIdx.x;
  const float* xp = &x[(size_t)row * D_];
  const float* rp = &r[(size_t)row * D_];

  float vals[4];
  float sum = 0.0f;
#pragma unroll
  for (int j = 0; j < 4; j++) {
    int c   = tid + j * 256;
    float v = xp[c] + rp[c];
    vals[j] = v;
    sum += v;
  }
#pragma unroll
  for (int off = 1; off < 32; off <<= 1) sum += __shfl_xor(sum, off, 32);
  if ((tid & 31) == 0) red[tid >> 5] = sum;
  __syncthreads();
  float tot = 0.0f;
#pragma unroll
  for (int w = 0; w < 8; w++) tot += red[w];
  float mean = tot * (1.0f / D_);
  __syncthreads();

  float vs = 0.0f;
#pragma unroll
  for (int j = 0; j < 4; j++) {
    float d = vals[j] - mean;
    vs += d * d;
  }
#pragma unroll
  for (int off = 1; off < 32; off <<= 1) vs += __shfl_xor(vs, off, 32);
  if ((tid & 31) == 0) red[tid >> 5] = vs;
  __syncthreads();
  float vtot = 0.0f;
#pragma unroll
  for (int w = 0; w < 8; w++) vtot += red[w];
  float rstd = rsqrtf(vtot * (1.0f / D_) + 1e-5f);

#pragma unroll
  for (int j = 0; j < 4; j++) {
    int c   = tid + j * 256;
    float y = (vals[j] - mean) * rstd * g[c] + bta[c];
    if (outF) outF[(size_t)row * D_ + c] = y;
    if (outB) outB[(size_t)row * D_ + c] = f2bf(y);
  }
}

// ---------------------------------------------------------------------------
// Orchestration
// ---------------------------------------------------------------------------
extern "C" void kernel_launch(void* const* d_in, const int* in_sizes, int n_in,
                              void* d_out, int out_size, void* d_ws,
                              size_t ws_size, hipStream_t stream) {
  (void)in_sizes; (void)n_in; (void)out_size; (void)ws_size;
  const int M = B_ * S_;  // 4096 rows

  const float* x   = (const float*)d_in[0];
  const float* enc = (const float*)d_in[1];
  // d_in[2]=tgt_mask (causal, analytic), d_in[3]=memory_mask (all ones)
  const float* sa_w[4] = {(const float*)d_in[4], (const float*)d_in[6],
                          (const float*)d_in[8], (const float*)d_in[10]};
  const float* sa_b[4] = {(const float*)d_in[5], (const float*)d_in[7],
                          (const float*)d_in[9], (const float*)d_in[11]};
  const float* ca_w[4] = {(const float*)d_in[12], (const float*)d_in[14],
                          (const float*)d_in[16], (const float*)d_in[18]};
  const float* ca_b[4] = {(const float*)d_in[13], (const float*)d_in[15],
                          (const float*)d_in[17], (const float*)d_in[19]};
  const float* ff_w1 = (const float*)d_in[20];
  const float* ff_b1 = (const float*)d_in[21];
  const float* ff_w2 = (const float*)d_in[22];
  const float* ff_b2 = (const float*)d_in[23];
  const float* ln_g[3] = {(const float*)d_in[24], (const float*)d_in[26],
                          (const float*)d_in[28]};
  const float* ln_b[3] = {(const float*)d_in[25], (const float*)d_in[27],
                          (const float*)d_in[29]};
  float* out = (float*)d_out;

  // ---- workspace carve-up ----
  char* ws   = (char*)d_ws;
  size_t off = 0;
  auto alloc = [&](size_t bytes) -> void* {
    void* p = ws + off;
    off     = (off + bytes + 255) & ~(size_t)255;
    return p;
  };
  bf16* xb   = (bf16*)alloc((size_t)M * D_ * 2);
  bf16* encb = (bf16*)alloc((size_t)M * D_ * 2);
  bf16* wsa[4];
  for (int i = 0; i < 4; i++) wsa[i] = (bf16*)alloc((size_t)D_ * D_ * 2);
  bf16* wca[4];
  for (int i = 0; i < 4; i++) wca[i] = (bf16*)alloc((size_t)D_ * D_ * 2);
  bf16* w1t    = (bf16*)alloc((size_t)D_ * DFF_ * 2);  // [DFF][D]
  bf16* w2t    = (bf16*)alloc((size_t)DFF_ * D_ * 2);  // [D][DFF]
  bf16* qb     = (bf16*)alloc((size_t)M * D_ * 2);
  bf16* kb     = (bf16*)alloc((size_t)M * D_ * 2);
  bf16* vbT    = (bf16*)alloc((size_t)M * D_ * 2);  // [B][H][DK][S]
  bf16* attnb  = (bf16*)alloc((size_t)M * D_ * 2);
  float* projF = (float*)alloc((size_t)M * D_ * 4);
  float* x1f   = (float*)alloc((size_t)M * D_ * 4);
  bf16* x1b    = (bf16*)alloc((size_t)M * D_ * 2);
  float* x2f   = (float*)alloc((size_t)M * D_ * 4);
  bf16* x2b    = (bf16*)alloc((size_t)M * D_ * 2);
  bf16* hb     = (bf16*)alloc((size_t)M * DFF_ * 2);

  auto cvt = [&](const float* src, bf16* dst, size_t n) {
    int blocks = (int)((n + 1023) / 1024);
    cvt_f32_bf16<<<dim3(blocks), dim3(256), 0, stream>>>(src, dst, (int)n);
  };
  auto cvtT = [&](const float* src, bf16* dst, int R, int C) {
    cvt_transpose_bf16<<<dim3(C / 32, R / 32), dim3(256), 0, stream>>>(src,
                                                                       dst, R,
                                                                       C);
  };
  auto lnorm = [&](const float* xi, const float* ri, int i, float* oF,
                   bf16* oB) {
    add_ln<<<dim3(M), dim3(256), 0, stream>>>(xi, ri, ln_g[i], ln_b[i], oF,
                                              oB);
  };

  // ---- stage 0: precision conversion (weights transposed to [N][K]) ----
  cvt(x, xb, (size_t)M * D_);
  cvt(enc, encb, (size_t)M * D_);
  for (int i = 0; i < 4; i++) {
    cvtT(sa_w[i], wsa[i], D_, D_);
    cvtT(ca_w[i], wca[i], D_, D_);
  }
  cvtT(ff_w1, w1t, D_, DFF_);
  cvtT(ff_w2, w2t, DFF_, D_);

  const dim3 agrid(S_ / 64, H_, B_);
  const dim3 gD(D_ / 128, M / 128);      // N=1024 projections
  const dim3 gF1(DFF_ / 128, M / 128);   // FFN up
  const dim3 gF2(D_ / 128, M / 128);     // FFN down

  // ---- self-attention (causal) ----
  gemm_bf16_wmma<0, 0, 1, 0><<<gD, 256, 0, stream>>>(xb, wsa[0], sa_b[0],
                                                     nullptr, qb, M, D_, D_);
  gemm_bf16_wmma<0, 0, 1, 0><<<gD, 256, 0, stream>>>(xb, wsa[1], sa_b[1],
                                                     nullptr, kb, M, D_, D_);
  gemm_bf16_wmma<0, 0, 0, 1><<<gD, 256, 0, stream>>>(xb, wsa[2], sa_b[2],
                                                     nullptr, vbT, M, D_, D_);
  attn_wmma<1><<<agrid, 128, 0, stream>>>(qb, kb, vbT, attnb, S_, S_);
  gemm_bf16_wmma<0, 1, 0, 0><<<gD, 256, 0, stream>>>(attnb, wsa[3], sa_b[3],
                                                     projF, nullptr, M, D_,
                                                     D_);
  lnorm(x, projF, 0, x1f, x1b);

  // ---- cross-attention (no mask) ----
  gemm_bf16_wmma<0, 0, 1, 0><<<gD, 256, 0, stream>>>(x1b, wca[0], ca_b[0],
                                                     nullptr, qb, M, D_, D_);
  gemm_bf16_wmma<0, 0, 1, 0><<<gD, 256, 0, stream>>>(encb, wca[1], ca_b[1],
                                                     nullptr, kb, M, D_, D_);
  gemm_bf16_wmma<0, 0, 0, 1><<<gD, 256, 0, stream>>>(encb, wca[2], ca_b[2],
                                                     nullptr, vbT, M, D_, D_);
  attn_wmma<0><<<agrid, 128, 0, stream>>>(qb, kb, vbT, attnb, S_, S_);
  gemm_bf16_wmma<0, 1, 0, 0><<<gD, 256, 0, stream>>>(attnb, wca[3], ca_b[3],
                                                     projF, nullptr, M, D_,
                                                     D_);
  lnorm(x1f, projF, 1, x2f, x2b);

  // ---- FFN ----
  gemm_bf16_wmma<1, 0, 1, 0><<<gF1, 256, 0, stream>>>(x2b, w1t, ff_b1,
                                                      nullptr, hb, M, DFF_,
                                                      D_);
  gemm_bf16_wmma<0, 1, 0, 0><<<gF2, 256, 0, stream>>>(hb, w2t, ff_b2, projF,
                                                      nullptr, M, D_, DFF_);
  lnorm(x2f, projF, 2, out, nullptr);
}
